// JointEmbeddingGATv2NBNet_8564164788938
// MI455X (gfx1250) — matrix operand order repaired and checked by reading, compiled, not run on hardware
//
#include <hip/hip_runtime.h>
#include <hip/hip_bf16.h>
#include <math.h>

// ---------------------------------------------------------------------------
// JointEmbeddingGATv2NBNet on MI455X (gfx1250, wave32, WMMA).
// Bandwidth-bound problem (~205MB output, ~13 GFLOP of GEMM); GEMMs run on
// v_wmma_f32_16x16x32_f16, edge phase uses L2-resident float atomics.
// ---------------------------------------------------------------------------

typedef __attribute__((ext_vector_type(16))) _Float16 v16h;
typedef __attribute__((ext_vector_type(8)))  _Float16 v8h;
typedef __attribute__((ext_vector_type(8)))  float    v8f;

#define EMBED 256
#define HIDDEN 128
#define OUTD 512
#define NCT 32
#define NEG_SLOPE 0.2f

__device__ __forceinline__ float elu_f(float v) {
    return v > 0.f ? v : (__expf(v) - 1.f);
}

__device__ __forceinline__ void atomicMaxFloat(float* addr, float value) {
    // Sign-split trick: positive floats order as ints, negative as reversed uints.
    if (value >= 0.f) atomicMax((int*)addr, __float_as_int(value));
    else              atomicMin((unsigned int*)addr, __float_as_uint(value));
}

// --- prep: transpose fp32 [K,Nn] weight into f16 [Nn,K] ---------------------
__global__ void k_prep_w(const float* __restrict__ W, _Float16* __restrict__ Wt,
                         int K, int Nn) {
    int idx = blockIdx.x * blockDim.x + threadIdx.x;
    if (idx >= K * Nn) return;
    int k = idx / Nn, n = idx % Nn;
    Wt[n * K + k] = (_Float16)W[k * Nn + n];
}

// --- prep: softplus table for dispersion [512,32] ---------------------------
__global__ void k_softplus(const float* __restrict__ d, float* __restrict__ sp, int total) {
    int i = blockIdx.x * blockDim.x + threadIdx.x;
    if (i >= total) return;
    float x = d[i];
    sp[i] = (x > 20.f) ? x : log1pf(__expf(x));
}

// --- embed: x = elu(W_embed[ct] + lipid*W_embed[32] + b); store f16 + ct ----
__global__ void k_embed(const float* __restrict__ onehot, const float* __restrict__ lipid,
                        const float* __restrict__ W_embed, const float* __restrict__ b_embed,
                        _Float16* __restrict__ x16, int* __restrict__ ct_idx, int N) {
    int i = blockIdx.x;
    if (i >= N) return;
    __shared__ int ct;
    int t = threadIdx.x;
    if (t < NCT) {
        if (onehot[(size_t)i * NCT + t] > 0.5f) ct = t;
    }
    __syncthreads();
    int c = ct;
    float lip = lipid[i];
    float v = W_embed[c * EMBED + t] + lip * W_embed[NCT * EMBED + t] + b_embed[t];
    x16[(size_t)i * EMBED + t] = (_Float16)elu_f(v);
    if (t == 0) ct_idx[i] = c;
}

// --- init m=-inf, denom=0, agg=0 -------------------------------------------
__global__ void k_init(float* __restrict__ m, float* __restrict__ denom,
                       float* __restrict__ agg, int N) {
    int i = blockIdx.x * blockDim.x + threadIdx.x;
    if (i < N * HIDDEN) agg[i] = 0.f;
    if (i < N) { m[i] = __int_as_float(0xff800000); denom[i] = 0.f; }
}

// --- WMMA GEMM: xl = x@W_l + b_l ; xr = x@W_r + b_r  (K=256, Ncols=128) -----
// One wave = one 16x16 output tile; 8 waves/block cover all 128 columns.
__global__ void k_gemm_lr(const _Float16* __restrict__ X,
                          const _Float16* __restrict__ Wl_t, const float* __restrict__ bl,
                          const _Float16* __restrict__ Wr_t, const float* __restrict__ br,
                          float* __restrict__ xl, float* __restrict__ xr, int N) {
    const _Float16* Wt = blockIdx.y ? Wr_t : Wl_t;
    const float* bias  = blockIdx.y ? br   : bl;
    float* out         = blockIdx.y ? xr   : xl;

    int wave = threadIdx.x >> 5;
    int lane = threadIdx.x & 31;
    int lo = lane & 15, hi = lane >> 4;
    int m0 = blockIdx.x * 16;
    int n0 = wave * 16;

    int row = m0 + lo; if (row >= N) row = N - 1;
    const _Float16* arow = X  + (size_t)row * EMBED;
    const _Float16* bcol = Wt + (size_t)(n0 + lo) * EMBED;

    v8f c = {};
#pragma unroll
    for (int kk = 0; kk < EMBED; kk += 32) {
        v8h a0 = *(const v8h*)(arow + kk + hi * 8);
        v8h a1 = *(const v8h*)(arow + kk + 16 + hi * 8);
        v8h b0 = *(const v8h*)(bcol + kk + hi * 8);
        v8h b1 = *(const v8h*)(bcol + kk + 16 + hi * 8);
        v16h a, b;
#pragma unroll
        for (int t = 0; t < 8; ++t) { a[t] = a0[t]; a[t + 8] = a1[t];
                                      b[t] = b0[t]; b[t + 8] = b1[t]; }
        c = __builtin_amdgcn_wmma_f32_16x16x32_f16(false, a, false, b,
                                                   (short)0, c, false, false);
    }
    float bn = bias[n0 + lo];
#pragma unroll
    for (int v = 0; v < 8; ++v) {
        int r = m0 + 8 * hi + v;
        if (r < N) out[(size_t)r * HIDDEN + n0 + lo] = c[v] + bn;
    }
}

// --- edge logits: wave per edge; logit = att . leaky_relu(xl[s]+xr[d]) ------
__global__ void k_edge_logit(const int* __restrict__ src, const int* __restrict__ dst,
                             const float* __restrict__ xl, const float* __restrict__ xr,
                             const float* __restrict__ att,
                             float* __restrict__ logit, float* __restrict__ mbuf, int E) {
    int e = blockIdx.x * 8 + (threadIdx.x >> 5);
    if (e >= E) return;
    int lane = threadIdx.x & 31;
    int s = src[e], d = dst[e];
    float4 l4 = *(const float4*)(xl + (size_t)s * HIDDEN + lane * 4);
    float4 r4 = *(const float4*)(xr + (size_t)d * HIDDEN + lane * 4);
    float4 a4 = *(const float4*)(att + lane * 4);
    float sum = 0.f, v;
    v = l4.x + r4.x; sum += (v > 0.f ? v : NEG_SLOPE * v) * a4.x;
    v = l4.y + r4.y; sum += (v > 0.f ? v : NEG_SLOPE * v) * a4.y;
    v = l4.z + r4.z; sum += (v > 0.f ? v : NEG_SLOPE * v) * a4.z;
    v = l4.w + r4.w; sum += (v > 0.f ? v : NEG_SLOPE * v) * a4.w;
#pragma unroll
    for (int off = 16; off > 0; off >>= 1) sum += __shfl_xor(sum, off, 32);
    if (lane == 0) { logit[e] = sum; atomicMaxFloat(mbuf + d, sum); }
}

// --- edge aggregate: unnormalized softmax accumulation ----------------------
__global__ void k_edge_agg(const int* __restrict__ src, const int* __restrict__ dst,
                           const float* __restrict__ xl, const float* __restrict__ logit,
                           const float* __restrict__ mbuf,
                           float* __restrict__ denom, float* __restrict__ agg, int E) {
    int e = blockIdx.x * 8 + (threadIdx.x >> 5);
    if (e >= E) return;
    int lane = threadIdx.x & 31;
    int s = src[e], d = dst[e];
    float alpha = __expf(logit[e] - mbuf[d]);
    if (lane == 0) atomicAdd(denom + d, alpha);
    float4 l4 = *(const float4*)(xl + (size_t)s * HIDDEN + lane * 4);
    float* ap = agg + (size_t)d * HIDDEN + lane * 4;
    atomicAdd(ap + 0, alpha * l4.x);
    atomicAdd(ap + 1, alpha * l4.y);
    atomicAdd(ap + 2, alpha * l4.z);
    atomicAdd(ap + 3, alpha * l4.w);
}

// --- h = elu(agg/denom + gat_bias), stored f16 ------------------------------
__global__ void k_hnode(const float* __restrict__ agg, const float* __restrict__ denom,
                        const float* __restrict__ gbias, _Float16* __restrict__ h16, int N) {
    int idx = blockIdx.x * blockDim.x + threadIdx.x;
    if (idx >= N * HIDDEN) return;
    int i = idx >> 7, j = idx & (HIDDEN - 1);
    float v = agg[idx] / (denom[i] + 1e-16f) + gbias[j];
    h16[idx] = (_Float16)elu_f(v);
}

// --- WMMA GEMM: logits = h@W_out + b_out (K=128, Ncols=512) -----------------
__global__ void k_gemm_out(const _Float16* __restrict__ H,
                           const _Float16* __restrict__ Wo_t, const float* __restrict__ bo,
                           float* __restrict__ out, int N) {
    int wave = threadIdx.x >> 5;
    int lane = threadIdx.x & 31;
    int lo = lane & 15, hi = lane >> 4;
    int m0 = blockIdx.x * 16;
    int n0 = (blockIdx.y * 8 + wave) * 16;

    int row = m0 + lo; if (row >= N) row = N - 1;
    const _Float16* arow = H    + (size_t)row * HIDDEN;
    const _Float16* bcol = Wo_t + (size_t)(n0 + lo) * HIDDEN;

    v8f c = {};
#pragma unroll
    for (int kk = 0; kk < HIDDEN; kk += 32) {
        v8h a0 = *(const v8h*)(arow + kk + hi * 8);
        v8h a1 = *(const v8h*)(arow + kk + 16 + hi * 8);
        v8h b0 = *(const v8h*)(bcol + kk + hi * 8);
        v8h b1 = *(const v8h*)(bcol + kk + 16 + hi * 8);
        v16h a, b;
#pragma unroll
        for (int t = 0; t < 8; ++t) { a[t] = a0[t]; a[t + 8] = a1[t];
                                      b[t] = b0[t]; b[t + 8] = b1[t]; }
        c = __builtin_amdgcn_wmma_f32_16x16x32_f16(false, a, false, b,
                                                   (short)0, c, false, false);
    }
    float bn = bo[n0 + lo];
#pragma unroll
    for (int v = 0; v < 8; ++v) {
        int r = m0 + 8 * hi + v;
        if (r < N) out[(size_t)r * OUTD + n0 + lo] = c[v] + bn;
    }
}

// --- disp gather: out[i,j] = softplus_table[j*32 + ct[i]] -------------------
__global__ void k_disp(const int* __restrict__ ct, const float* __restrict__ sp,
                       float* __restrict__ out, int N) {
    int idx = blockIdx.x * blockDim.x + threadIdx.x;
    if (idx >= N * OUTD) return;
    int i = idx >> 9, j = idx & (OUTD - 1);
    out[idx] = sp[j * NCT + ct[i]];
}

extern "C" void kernel_launch(void* const* d_in, const int* in_sizes, int n_in,
                              void* d_out, int out_size, void* d_ws, size_t ws_size,
                              hipStream_t stream) {
    const float* onehot  = (const float*)d_in[0];
    const float* lipid   = (const float*)d_in[1];
    const int*   edge    = (const int*)  d_in[2];
    const float* W_embed = (const float*)d_in[3];
    const float* b_embed = (const float*)d_in[4];
    const float* W_l     = (const float*)d_in[5];
    const float* b_l     = (const float*)d_in[6];
    const float* W_r     = (const float*)d_in[7];
    const float* b_r     = (const float*)d_in[8];
    const float* att     = (const float*)d_in[9];
    const float* gbias   = (const float*)d_in[10];
    const float* W_out   = (const float*)d_in[11];
    const float* b_out   = (const float*)d_in[12];
    const float* disper  = (const float*)d_in[13];

    const int N = in_sizes[1];
    const int E = in_sizes[2] / 2;
    const int* src = edge;
    const int* dst = edge + E;

    // Carve workspace (256B aligned chunks).
    char* p = (char*)d_ws;
    auto carve = [&](size_t bytes) -> char* {
        char* r = p; p += (bytes + 255) & ~(size_t)255; return r;
    };
    int*      ct_idx = (int*)     carve((size_t)N * 4);
    _Float16* x16    = (_Float16*)carve((size_t)N * EMBED * 2);
    float*    xl     = (float*)   carve((size_t)N * HIDDEN * 4);
    float*    xr     = (float*)   carve((size_t)N * HIDDEN * 4);
    float*    agg    = (float*)   carve((size_t)N * HIDDEN * 4);
    float*    mbuf   = (float*)   carve((size_t)N * 4);
    float*    denom  = (float*)   carve((size_t)N * 4);
    float*    logit  = (float*)   carve((size_t)E * 4);
    _Float16* h16    = (_Float16*)carve((size_t)N * HIDDEN * 2);
    _Float16* Wl_t   = (_Float16*)carve((size_t)EMBED * HIDDEN * 2);
    _Float16* Wr_t   = (_Float16*)carve((size_t)EMBED * HIDDEN * 2);
    _Float16* Wo_t   = (_Float16*)carve((size_t)HIDDEN * OUTD * 2);
    float*    sp     = (float*)   carve((size_t)OUTD * NCT * 4);

    float* out_logits = (float*)d_out;
    float* out_disp   = out_logits + (size_t)N * OUTD;

    // Weight prep (tiny).
    k_prep_w<<<(EMBED * HIDDEN + 255) / 256, 256, 0, stream>>>(W_l, Wl_t, EMBED, HIDDEN);
    k_prep_w<<<(EMBED * HIDDEN + 255) / 256, 256, 0, stream>>>(W_r, Wr_t, EMBED, HIDDEN);
    k_prep_w<<<(HIDDEN * OUTD + 255) / 256, 256, 0, stream>>>(W_out, Wo_t, HIDDEN, OUTD);
    k_softplus<<<(OUTD * NCT + 255) / 256, 256, 0, stream>>>(disper, sp, OUTD * NCT);

    // Embed (one-hot gather + elu), ct extraction.
    k_embed<<<N, EMBED, 0, stream>>>(onehot, lipid, W_embed, b_embed, x16, ct_idx, N);

    // Init softmax state.
    k_init<<<(N * HIDDEN + 255) / 256, 256, 0, stream>>>(mbuf, denom, agg, N);

    // x_l / x_r via WMMA.
    dim3 gLR((N + 15) / 16, 2);
    k_gemm_lr<<<gLR, 256, 0, stream>>>(x16, Wl_t, b_l, Wr_t, b_r, xl, xr, N);

    // Edge phase (L2-resident atomics).
    k_edge_logit<<<(E + 7) / 8, 256, 0, stream>>>(src, dst, xl, xr, att, logit, mbuf, E);
    k_edge_agg<<<(E + 7) / 8, 256, 0, stream>>>(src, dst, xl, logit, mbuf, denom, agg, E);

    // h = elu(agg/denom + bias), f16.
    k_hnode<<<(N * HIDDEN + 255) / 256, 256, 0, stream>>>(agg, denom, gbias, h16, N);

    // Output GEMM via WMMA.
    dim3 gO((N + 15) / 16, OUTD / (8 * 16));
    k_gemm_out<<<gO, 256, 0, stream>>>(h16, Wo_t, b_out, out_logits, N);

    // Dispersion gather.
    k_disp<<<(N * OUTD + 255) / 256, 256, 0, stream>>>(ct_idx, sp, out_disp, N);
}
